// TriangleAttention_50079318671462
// MI455X (gfx1250) — matrix operand order repaired and checked by reading, not compiled, not run on hardware
//
#include <hip/hip_runtime.h>
#include <hip/hip_bf16.h>
#include <stdint.h>

// ---------------------------------------------------------------------------
// TriangleAttention for MI455X (gfx1250, wave32, WMMA 16x16x32 bf16)
// B=1, S=192, R=192, D=256, H=8, DH=32
// ---------------------------------------------------------------------------

typedef __attribute__((ext_vector_type(16))) __bf16 v16bf;
typedef __attribute__((ext_vector_type(8)))  __bf16 v8bf;
typedef __attribute__((ext_vector_type(8)))  float  v8f;

#define SDIM 192
#define RDIM 192
#define DDIM 256
#define HNUM 8
#define QKV3 768
#define NROW (SDIM * RDIM)          // 36864 (s,i) positions
#define SQRT_DH 5.656854249492381f  // sqrt(32): reference MULTIPLIES by sqrt(key_dim)
#define MASK_BIAS -1.0e9f

__device__ __forceinline__ v8bf ld8(const __bf16* p) { return *(const v8bf*)p; }

__device__ __forceinline__ v16bf mk16(v8bf lo, v8bf hi) {
  v16bf a;
#pragma unroll
  for (int i = 0; i < 8; i++) { a[i] = lo[i]; a[i + 8] = hi[i]; }
  return a;
}

__device__ __forceinline__ v8f wmma_bf16(v16bf a, v16bf b, v8f c) {
  // (neg_a, A, neg_b, B, c_mod, C, reuse_a, reuse_b)
  return __builtin_amdgcn_wmma_f32_16x16x32_bf16(false, a, false, b, (short)0, c,
                                                 false, false);
}

// ---------------------------------------------------------------------------
// Kernel 1: LayerNorm (eps=1e-6) over D=256, emit bf16 activations.
// One wave per row; each lane owns 8 consecutive floats.
// ---------------------------------------------------------------------------
__global__ void __launch_bounds__(256) ln_bf16_kernel(
    const float* __restrict__ x, const float* __restrict__ g,
    const float* __restrict__ b, __bf16* __restrict__ xln) {
  int row  = blockIdx.x * 8 + (threadIdx.x >> 5);
  int lane = threadIdx.x & 31;
  const float* xr = x + (size_t)row * DDIM + lane * 8;
  float v[8];
#pragma unroll
  for (int i = 0; i < 8; i++) v[i] = xr[i];
  float s = 0.f, s2 = 0.f;
#pragma unroll
  for (int i = 0; i < 8; i++) { s += v[i]; s2 += v[i] * v[i]; }
#pragma unroll
  for (int off = 1; off < 32; off <<= 1) {
    s  += __shfl_xor(s, off, 32);
    s2 += __shfl_xor(s2, off, 32);
  }
  float mu  = s * (1.f / DDIM);
  float var = s2 * (1.f / DDIM) - mu * mu;
  float inv = rsqrtf(var + 1e-6f);
  v8bf y;
#pragma unroll
  for (int i = 0; i < 8; i++) {
    int c = lane * 8 + i;
    y[i] = (__bf16)((v[i] - mu) * inv * g[c] + b[c]);
  }
  *(v8bf*)(xln + (size_t)row * DDIM + lane * 8) = y;
}

// ---------------------------------------------------------------------------
// Kernel 2: weights fp32 -> bf16 (Wqkv 768x256, Wout 256x256)
// ---------------------------------------------------------------------------
__global__ void __launch_bounds__(256) cvt_w_kernel(
    const float* __restrict__ wqkv, const float* __restrict__ wout,
    __bf16* __restrict__ oq, __bf16* __restrict__ oo) {
  int i = blockIdx.x * 256 + threadIdx.x;
  const int NQ = QKV3 * DDIM;  // 196608
  if (i < NQ) oq[i] = (__bf16)wqkv[i];
  else        oo[i - NQ] = (__bf16)wout[i - NQ];
}

// ---------------------------------------------------------------------------
// Kernel 3/5: GEMM  out[m,n] = sum_k A[m,k] * W[n,k] + bias[n]
// A: bf16 [M,256] row-major, W: bf16 [N,256] row-major.
// One 16x64 output strip per wave: 4 C-frags share each A-frag load
// (4 WMMAs per K-step, K-loop of 8 steps).
// Fragment layouts per CDNA5 ISA (wave32):
//   A 16x32: lane L -> row L%16; e<8: k = (L/16)*8+e ; e>=8: k = 16+(L/16)*8+(e-8)
//   B 32x16: lane L -> col L%16; e<8: k = (L/16)*8+e ; e>=8: k = 16+(L/16)*8+(e-8)
//   C 16x16: vgpr r, lane L -> (row r + 8*(L/16), col L%16)
// ---------------------------------------------------------------------------
template <bool OUTBF>
__global__ void __launch_bounds__(256) gemm_bf16_kernel(
    const __bf16* __restrict__ A, const __bf16* __restrict__ W,
    const float* __restrict__ bias, void* __restrict__ outp, int nquad,
    int ldo) {
  int lane = threadIdx.x & 31, wv = threadIdx.x >> 5;
  int tile = blockIdx.x * 8 + wv;
  int mt = tile / nquad, ng = tile % nquad;
  int col = lane & 15, half = lane >> 4;
  int m0 = mt * 16;
  const __bf16* arow = A + (size_t)(m0 + col) * DDIM;
  const __bf16* wr[4];
#pragma unroll
  for (int u = 0; u < 4; u++)
    wr[u] = W + (size_t)(ng * 64 + u * 16 + col) * DDIM;
  v8f c[4] = {};
#pragma unroll
  for (int kb = 0; kb < DDIM / 32; kb++) {
    const __bf16* ap = arow + kb * 32 + half * 8;
    v16bf af = mk16(ld8(ap), ld8(ap + 16));
#pragma unroll
    for (int u = 0; u < 4; u++) {
      const __bf16* wp = wr[u] + kb * 32 + half * 8;
      v16bf wf = mk16(ld8(wp), ld8(wp + 16));
      c[u] = wmma_bf16(af, wf, c[u]);
    }
  }
#pragma unroll
  for (int u = 0; u < 4; u++) {
    int n = ng * 64 + u * 16 + col;
    float bn = bias[n];
#pragma unroll
    for (int r = 0; r < 8; r++) {
      size_t m = (size_t)(m0 + r + 8 * half);
      float val = c[u][r] + bn;
      if (OUTBF) ((__bf16*)outp)[m * ldo + n] = (__bf16)val;
      else       ((float*)outp)[m * ldo + n] = val;
    }
  }
}

// ---------------------------------------------------------------------------
// Kernel 4: attention per (s,h). Block = 128 threads (4 waves),
// each wave owns 3 query row-tiles of 16.
//  - K tile [192][32] staged to LDS with CDNA5 async-to-LDS b128
//    (ASYNCcnt-tracked), V tile staged TRANSPOSED [32][192] manually.
//  - 12 score C-frags in registers, softmax via half-wave shuffles,
//    post-softmax multiplicative bias, C->A transpose through LDS,
//    then 6x2 WMMAs for A.V with contiguous ds_load_b128 B-frags.
// qkv layout: row (s*192+i)*768, q at +0, k at +256, v at +512, per head h*32.
// ---------------------------------------------------------------------------
struct AttnSmem {
  __bf16 kbuf[RDIM * 32];     // 12 KB, [j][dh]
  __bf16 vt[32 * RDIM];       // 12 KB, [dh][j]  (transposed V)
  __bf16 wbuf[4][16][RDIM];   // 24 KB, per-wave softmax weights
};

__global__ void __launch_bounds__(128) attn_kernel(
    const __bf16* __restrict__ qkv, const int* __restrict__ mask,
    const float* __restrict__ bias, __bf16* __restrict__ attnout) {
  __shared__ __align__(16) AttnSmem sm;

  int lane = threadIdx.x & 31, wv = threadIdx.x >> 5;
  int s = blockIdx.x >> 3, h = blockIdx.x & 7;
  int col = lane & 15, half = lane >> 4;

  const __bf16* qbase = qkv + (size_t)s * RDIM * QKV3 + h * 32;

  // ---- cooperative staging: K via async-to-LDS, V^T via manual scatter ----
#pragma unroll
  for (int itc = 0; itc < 6; itc++) {
    int c = itc * 128 + threadIdx.x;  // 768 16B-chunks: j = c/4, part p = c%4
    int j = c >> 2, p = c & 3;
    const __bf16* gk = qbase + (size_t)j * QKV3 + 256 + p * 8;
    uint32_t loff = (uint32_t)((j * 32 + p * 8) * 2);  // kbuf at struct offset 0
    uint64_t ga = (uint64_t)(uintptr_t)gk;
    asm volatile("global_load_async_to_lds_b128 %0, %1, off"
                 :: "v"(loff), "v"(ga) : "memory");
    v8bf vv = ld8(gk + 256);  // V row j, d = p*8..p*8+7
#pragma unroll
    for (int e = 0; e < 8; e++) sm.vt[(p * 8 + e) * RDIM + j] = vv[e];
  }
  asm volatile("s_wait_asynccnt 0" ::: "memory");
  __syncthreads();

  for (int t = 0; t < 3; t++) {
    int it = wv * 3 + t;  // query tile 0..11
    // --- Q fragment (A layout, K = DH = 32 in one WMMA) ---
    const __bf16* qr = qbase + (size_t)(it * 16 + col) * QKV3 + half * 8;
    v16bf aq = mk16(ld8(qr), ld8(qr + 16));

    // --- scores: 12 tiles of 16x16 (K B-frags from LDS), masked + scaled ---
    v8f sc[12];
#pragma unroll
    for (int jt = 0; jt < 12; jt++) {
      int j = jt * 16 + col;
      const __bf16* kp = sm.kbuf + j * 32 + half * 8;
      v16bf bk = mk16(ld8(kp), ld8(kp + 16));
      v8f c = {};
      c = wmma_bf16(aq, bk, c);
      bool msk = mask[s * RDIM + j] != 0;  // column-only mask -> per-lane bool
      v8f r;
#pragma unroll
      for (int q = 0; q < 8; q++) r[q] = msk ? MASK_BIAS : c[q] * SQRT_DH;
      sc[jt] = r;
    }

    // --- row max (each row lives in one 16-lane half at fixed vgpr idx) ---
    v8f rmax = sc[0];
#pragma unroll
    for (int jt = 1; jt < 12; jt++)
#pragma unroll
      for (int q = 0; q < 8; q++) rmax[q] = fmaxf(rmax[q], sc[jt][q]);
#pragma unroll
    for (int off = 1; off < 16; off <<= 1)
#pragma unroll
      for (int q = 0; q < 8; q++)
        rmax[q] = fmaxf(rmax[q], __shfl_xor(rmax[q], off, 32));

    // --- exp + row sum ---
    v8f rsum = {};
#pragma unroll
    for (int jt = 0; jt < 12; jt++)
#pragma unroll
      for (int q = 0; q < 8; q++) {
        float e = __expf(sc[jt][q] - rmax[q]);
        sc[jt][q] = e;
        rsum[q] += e;
      }
#pragma unroll
    for (int off = 1; off < 16; off <<= 1)
#pragma unroll
      for (int q = 0; q < 8; q++) rsum[q] += __shfl_xor(rsum[q], off, 32);

    // --- normalize, post-softmax multiplicative bias, stash bf16 in LDS ---
#pragma unroll
    for (int q = 0; q < 8; q++) {
      int i = it * 16 + q + 8 * half;  // global query row
      const float* brow = bias + ((size_t)h * RDIM + i) * RDIM;
      float rinv = 1.f / rsum[q];
#pragma unroll
      for (int jt = 0; jt < 12; jt++) {
        int j = jt * 16 + col;
        sm.wbuf[wv][q + 8 * half][j] = (__bf16)(sc[jt][q] * rinv * brow[j]);
      }
    }
    // same-wave LDS write -> transposed read: make the ordering explicit
    asm volatile("s_wait_dscnt 0" ::: "memory");

    // --- A.V : out tile 16x32 = two 16x16 C frags, K = 192 in 6 steps ---
#pragma unroll
    for (int dt = 0; dt < 2; dt++) {
      v8f c = {};
      int d = dt * 16 + col;
#pragma unroll
      for (int kb = 0; kb < 6; kb++) {
        const __bf16* wr = &sm.wbuf[wv][col][kb * 32 + half * 8];
        v16bf aw = mk16(*(const v8bf*)wr, *(const v8bf*)(wr + 16));
        const __bf16* vp = sm.vt + d * RDIM + kb * 32 + half * 8;
        v16bf bv = mk16(ld8(vp), ld8(vp + 16));
        c = wmma_bf16(aw, bv, c);
      }
#pragma unroll
      for (int q = 0; q < 8; q++) {
        int i = it * 16 + q + 8 * half;
        attnout[((size_t)s * RDIM + i) * DDIM + (h * 32 + d)] = (__bf16)c[q];
      }
    }
  }
}

// ---------------------------------------------------------------------------
// Host-side orchestration
// ---------------------------------------------------------------------------
extern "C" void kernel_launch(void* const* d_in, const int* in_sizes, int n_in,
                              void* d_out, int out_size, void* d_ws,
                              size_t ws_size, hipStream_t stream) {
  (void)in_sizes; (void)n_in; (void)out_size; (void)ws_size;
  const float* pair_act = (const float*)d_in[0];
  const int*   pair_msk = (const int*)d_in[1];
  const float* bias     = (const float*)d_in[2];
  const float* ln_g     = (const float*)d_in[3];
  const float* ln_b     = (const float*)d_in[4];
  const float* Wqkv     = (const float*)d_in[5];
  const float* b_qkv    = (const float*)d_in[6];
  const float* Wout     = (const float*)d_in[7];
  const float* b_out    = (const float*)d_in[8];
  float* out = (float*)d_out;

  // workspace carve-up (all bf16), total ~95 MB (fits MI455X 192MB L2)
  __bf16* xln    = (__bf16*)d_ws;                       // 36864*256
  __bf16* qkvbf  = xln    + (size_t)NROW * DDIM;        // 36864*768
  __bf16* attnbf = qkvbf  + (size_t)NROW * QKV3;        // 36864*256
  __bf16* wqkvbf = attnbf + (size_t)NROW * DDIM;        // 768*256
  __bf16* woutbf = wqkvbf + (size_t)QKV3 * DDIM;        // 256*256

  // 1) LayerNorm -> bf16            (36864 rows, 8 rows/block)
  ln_bf16_kernel<<<NROW / 8, 256, 0, stream>>>(pair_act, ln_g, ln_b, xln);
  // 2) weight convert               (262144 elems)
  cvt_w_kernel<<<(QKV3 * DDIM + DDIM * DDIM) / 256, 256, 0, stream>>>(
      Wqkv, Wout, wqkvbf, woutbf);
  // 3) QKV GEMM: [36864,256] x [256,768] -> bf16 (16x64 strips per wave)
  gemm_bf16_kernel<true><<<(NROW / 16) * (QKV3 / 64) / 8, 256, 0, stream>>>(
      xln, wqkvbf, b_qkv, qkvbf, QKV3 / 64, QKV3);
  // 4) attention per (s,h)
  attn_kernel<<<SDIM * HNUM, 128, 0, stream>>>(qkvbf, pair_msk, bias, attnbf);
  // 5) out-proj GEMM: [36864,256] x [256,256] -> fp32 d_out
  gemm_bf16_kernel<false><<<(NROW / 16) * (DDIM / 64) / 8, 256, 0, stream>>>(
      attnbf, woutbf, b_out, out, DDIM / 64, DDIM);
}
